// LiquidNeuralNetwork_16844861735745
// MI455X (gfx1250) — matrix operand order repaired and checked by reading, not compiled, not run on hardware
//
#include <hip/hip_runtime.h>
#include <hip/hip_bf16.h>

typedef __bf16 v16bf __attribute__((ext_vector_type(16)));
typedef float  v8f   __attribute__((ext_vector_type(8)));

#define HDIM   512
#define IDIM   256
#define ODIM   128
#define BATCH  32768
#define TILE_B 64
#define NTHREADS 512
#define DT 0.1f
#define LN_EPS 1e-5f

// Workspace layout (offsets in bf16 elements)
#define SZ_WI0  (256*512)
#define SZ_WH   (512*512)
#define SZ_WO   (512*128)
#define OFF_WI0 0
#define OFF_WI1 (OFF_WI0+SZ_WI0)
#define OFF_WI2 (OFF_WI1+SZ_WH)
#define OFF_WR0 (OFF_WI2+SZ_WH)
#define OFF_WR1 (OFF_WR0+SZ_WH)
#define OFF_WR2 (OFF_WR1+SZ_WH)
#define OFF_WO  (OFF_WR2+SZ_WH)

// Dynamic LDS layout (bytes)
#define SM_HFRAG 0                    // 4*16*32*16 bf16 = 65536 B  (h, A-fragment layout)
#define SM_IILIN 65536                // 64*512 bf16     = 65536 B  (i_input, linear)
#define SM_HNEW  131072               // 4*32*8*32 f32   = 131072 B (h_new, C-fragment layout)
#define SM_VEC   262144               // 5*512 f32       = 10240 B
#define SM_STATS 272384               // 64*8*2 f32      = 4096 B
#define SM_MRV   276480               // 64*2 f32        = 512 B
#define SMEM_BYTES 276992

// ---------------------------------------------------------------------------
// Pack a K x N matrix (optionally transposed-from-src, optionally masked) into
// bf16 WMMA B-fragment layout: dst[((nt*KT + kt)*32 + lane)*16 + e]
// ---------------------------------------------------------------------------
__global__ void lnn_pack_b_kernel(const float* __restrict__ src,
                                  const float* __restrict__ mask,
                                  __bf16* __restrict__ dst,
                                  int K, int N, int transpose) {
    int idx = blockIdx.x * blockDim.x + threadIdx.x;
    if (idx >= K * N) return;
    int n = idx % N;
    int k = idx / N;
    float val = transpose ? src[(size_t)n * K + k] : src[(size_t)k * N + n];
    if (mask) val *= mask[(size_t)n * K + k];
    int KT   = K >> 5;
    int nt   = n >> 4;
    int kt   = k >> 5;
    int kk   = k & 31;
    int half = (kk >> 3) & 1;
    int lane = (n & 15) + 16 * half;
    int e    = (kk & 7) + 8 * (kk >> 4);
    dst[(((size_t)nt * KT + kt) * 32 + lane) * 16 + e] = (__bf16)val;
}

// A-fragment slot for element (row, col): [mt(4)][kt(16)][lane(32)][e(16)]
//   K = 16*(v>>2) + 2*(v&3) + 8*half + j,  e = 2v + j
__device__ __forceinline__ int frag_index(int row, int col) {
    int m    = row & 15;
    int kk   = col & 31;
    int half = (kk >> 3) & 1;
    int lane = m + 16 * half;
    int kkp  = kk - 8 * half;
    int v    = 4 * (kkp >> 4) + ((kkp & 7) >> 1);
    int j    = kk & 1;
    int e    = 2 * v + j;
    int mt   = row >> 4;
    int kt   = col >> 5;
    return (((mt * 16 + kt) * 32 + lane) * 16 + e);
}

// ---------------------------------------------------------------------------
// GEMM over NT 16x16 tiles sharing one M-tile.  KT is compile-time so all
// tile/k offsets become immediate fields; B fragments double-buffered so the
// loads for kt+1 are in flight while the NT WMMAs for kt execute.
// ---------------------------------------------------------------------------
template<int NT, int KT>
__device__ __forceinline__ void gemm_frag(v8f acc[NT],
                                          const __bf16* __restrict__ afrag,
                                          const __bf16* __restrict__ bpk,
                                          int mt, int nt0, int lane) {
    const char* abase = (const char*)afrag + mt * (16 * 1024) + lane * 32;
    const char* bbase = (const char*)bpk + (size_t)nt0 * (KT * 1024) + lane * 32;

    v16bf bcur[NT], bnxt[NT];
    #pragma unroll
    for (int t = 0; t < NT; ++t)
        bcur[t] = *(const v16bf*)(bbase + t * (KT * 1024));

    #pragma unroll
    for (int kt = 0; kt < KT; ++kt) {
        v16bf a = *(const v16bf*)(abase + kt * 1024);
        if (kt + 1 < KT) {
            #pragma unroll
            for (int t = 0; t < NT; ++t)
                bnxt[t] = *(const v16bf*)(bbase + (t * KT + kt + 1) * 1024);
        }
        #pragma unroll
        for (int t = 0; t < NT; ++t)
            acc[t] = __builtin_amdgcn_wmma_f32_16x16x32_bf16(
                false, a, false, bcur[t], (short)0, acc[t], false, false);
        if (kt + 1 < KT) {
            #pragma unroll
            for (int t = 0; t < NT; ++t) bcur[t] = bnxt[t];
        }
    }
}

template<int NT>
__device__ __forceinline__ void zero_acc(v8f acc[NT]) {
    #pragma unroll
    for (int t = 0; t < NT; ++t)
        #pragma unroll
        for (int i = 0; i < 8; ++i) acc[t][i] = 0.0f;
}

// ---------------------------------------------------------------------------
// Fused liquid network: per 64-row batch tile, run all 3 layers (input GEMM,
// 10 ODE steps with per-step LayerNorm) and the output projection, with h
// resident in LDS the whole time.
// ---------------------------------------------------------------------------
__global__ void __launch_bounds__(NTHREADS, 1)
lnn_fused_kernel(const float* __restrict__ x,
                 const float* __restrict__ bias0, const float* __restrict__ tau0,
                 const float* __restrict__ Av0,   const float* __restrict__ g0,
                 const float* __restrict__ be0,
                 const float* __restrict__ bias1, const float* __restrict__ tau1,
                 const float* __restrict__ Av1,   const float* __restrict__ g1,
                 const float* __restrict__ be1,
                 const float* __restrict__ bias2, const float* __restrict__ tau2,
                 const float* __restrict__ Av2,   const float* __restrict__ g2,
                 const float* __restrict__ be2,
                 const float* __restrict__ out_b,
                 const __bf16* __restrict__ wpk,
                 float* __restrict__ out) {
    extern __shared__ char smem[];
    __bf16* hfrag = (__bf16*)(smem + SM_HFRAG);
    __bf16* iiln  = (__bf16*)(smem + SM_IILIN);
    float*  hnew  = (float*) (smem + SM_HNEW);
    float*  vbias = (float*) (smem + SM_VEC);
    float*  vA    = vbias + HDIM;
    float*  vc    = vA + HDIM;
    float*  vg    = vc + HDIM;
    float*  vbe   = vg + HDIM;
    float*  stats = (float*) (smem + SM_STATS);
    float*  mrv   = (float*) (smem + SM_MRV);

    const int tid  = threadIdx.x;
    const int lane = tid & 31;
    const int wave = tid >> 5;
    const int b0   = blockIdx.x * TILE_B;

    const float* biasL[3] = {bias0, bias1, bias2};
    const float* tauL[3]  = {tau0, tau1, tau2};
    const float* AL[3]    = {Av0, Av1, Av2};
    const float* gL[3]    = {g0, g1, g2};
    const float* beL[3]   = {be0, be1, be2};
    const size_t wiOff[3] = {OFF_WI0, OFF_WI1, OFF_WI2};
    const size_t wrOff[3] = {OFF_WR0, OFF_WR1, OFF_WR2};

    const int mt   = wave >> 2;       // M-tile (0..3) owned by this wave
    const int nt0  = 8 * (wave & 3);  // first of 8 N-tiles owned by this wave
    const int half = lane >> 4;
    const int nn   = lane & 15;

    for (int layer = 0; layer < 3; ++layer) {
        // Per-column vectors for this layer (c = 1 + dt/tau precomputed).
        for (int i = tid; i < HDIM; i += NTHREADS) {
            vbias[i] = biasL[layer][i];
            vA[i]    = AL[layer][i];
            vc[i]    = 1.0f + DT / tauL[layer][i];
            vg[i]    = gL[layer][i];
            vbe[i]   = beL[layer][i];
        }
        if (layer == 0) {
            // Stage the x tile into A-fragment layout (bf16), kt 0..7.
            for (int idx = tid; idx < TILE_B * IDIM; idx += NTHREADS) {
                int row = idx >> 8, col = idx & 255;
                hfrag[frag_index(row, col)] =
                    (__bf16)x[(size_t)(b0 + row) * IDIM + col];
            }
        }
        __syncthreads();

        // i_input = X @ W_in   (X = x tile for layer 0, previous h otherwise)
        {
            v8f acc[8];
            zero_acc<8>(acc);
            if (layer == 0)
                gemm_frag<8, 8>(acc, hfrag, wpk + wiOff[0], mt, nt0, lane);
            else
                gemm_frag<8, 16>(acc, hfrag, wpk + wiOff[layer], mt, nt0, lane);
            #pragma unroll
            for (int t = 0; t < 8; ++t) {
                int col   = 16 * (nt0 + t) + nn;
                int ibase = (16 * mt + 8 * half) * HDIM + col;
                #pragma unroll
                for (int r = 0; r < 8; ++r)
                    iiln[ibase + r * HDIM] = (__bf16)acc[t][r];
            }
        }
        __syncthreads();
        // h0 = 0
        for (int i = tid; i < 4 * 16 * 32 * 16; i += NTHREADS)
            hfrag[i] = (__bf16)0.0f;
        __syncthreads();

        const __bf16* Wr = wpk + wrOff[layer];
        for (int step = 0; step < 10; ++step) {
            // S = h @ Wr_T (bf16 WMMA, f32 accumulate)
            v8f acc[8];
            zero_acc<8>(acc);
            gemm_frag<8, 16>(acc, hfrag, Wr, mt, nt0, lane);

            // Fused sigmoid / semi-implicit ODE update -> hnew (C-frag f32)
            #pragma unroll
            for (int t = 0; t < 8; ++t) {
                int nt  = nt0 + t;
                int col = 16 * nt + nn;
                float bcol = vbias[col];
                float Acol = vA[col];
                float ccol = vc[col];
                // h_old fragment index is affine in r: lane_a = r + 8*half + 16*halfA
                int kk    = col & 31;
                int halfA = (kk >> 3) & 1;
                int kkp   = kk - 8 * halfA;
                int vv    = 4 * (kkp >> 4) + ((kkp & 7) >> 1);
                int e     = 2 * vv + (kk & 1);
                int fbase = (((mt * 16 + (col >> 5)) * 32) +
                             (8 * half + 16 * halfA)) * 16 + e;
                int ibase = (16 * mt + 8 * half) * HDIM + col;
                int hbase = ((mt * 32 + nt) * 8) * 32 + lane;
                #pragma unroll
                for (int r = 0; r < 8; ++r) {
                    float hold = (float)hfrag[fbase + r * 16];
                    float pre  = acc[t][r] + (float)iiln[ibase + r * HDIM] + bcol;
                    float f    = 1.0f / (1.0f + __expf(-pre));
                    float hn   = (hold + DT * f * Acol) / (ccol + DT * f);
                    hnew[hbase + r * 32] = hn;
                }
            }
            __syncthreads();

            // LayerNorm stats: 8 partial segments of 64 columns per row.
            {
                int row  = tid >> 3;
                int part = tid & 7;
                int mt2 = row >> 4, r = row & 7, half2 = (row >> 3) & 1;
                float s = 0.0f, ss = 0.0f;
                for (int c = part * 64; c < part * 64 + 64; ++c) {
                    int nt    = c >> 4;
                    int lane2 = 16 * half2 + (c & 15);
                    float hv  = hnew[((mt2 * 32 + nt) * 8 + r) * 32 + lane2];
                    s += hv; ss += hv * hv;
                }
                stats[(row * 8 + part) * 2]     = s;
                stats[(row * 8 + part) * 2 + 1] = ss;
            }
            __syncthreads();
            if (tid < TILE_B) {
                float s = 0.0f, ss = 0.0f;
                for (int p = 0; p < 8; ++p) {
                    s  += stats[(tid * 8 + p) * 2];
                    ss += stats[(tid * 8 + p) * 2 + 1];
                }
                float mean = s * (1.0f / HDIM);
                float var  = ss * (1.0f / HDIM) - mean * mean;
                mrv[tid * 2]     = mean;
                mrv[tid * 2 + 1] = rsqrtf(var + LN_EPS);
            }
            __syncthreads();

            // Normalize and write h back into A-fragment layout (bf16).
            for (int slot = tid; slot < 4 * 16 * 32 * 16; slot += NTHREADS) {
                int e      = slot & 15;
                int lane_a = (slot >> 4) & 31;
                int kt     = (slot >> 9) & 15;
                int mt2    = slot >> 13;
                int m      = lane_a & 15;
                int halfA  = lane_a >> 4;
                int vv     = e >> 1;
                int j      = e & 1;
                int K      = 16 * (vv >> 2) + 2 * (vv & 3) + 8 * halfA + j;
                int row    = 16 * mt2 + m;
                int col    = 32 * kt + K;
                int r      = row & 7;
                int half2  = (row >> 3) & 1;
                int nt     = col >> 4;
                int lane2  = 16 * half2 + (col & 15);
                float hv = hnew[((mt2 * 32 + nt) * 8 + r) * 32 + lane2];
                float o  = (hv - mrv[row * 2]) * mrv[row * 2 + 1] * vg[col] + vbe[col];
                hfrag[slot] = (__bf16)o;
            }
            __syncthreads();
        }
    }

    // Output projection: out = h @ out_W.T + out_b   (O = 128 -> 2 tiles/wave)
    {
        int nt0o = 2 * (wave & 3);
        v8f acc[2];
        zero_acc<2>(acc);
        gemm_frag<2, 16>(acc, hfrag, wpk + OFF_WO, mt, nt0o, lane);
        #pragma unroll
        for (int t = 0; t < 2; ++t) {
            int col = 16 * (nt0o + t) + nn;
            float ob = out_b[col];
            #pragma unroll
            for (int r = 0; r < 8; ++r) {
                int row = 16 * mt + r + 8 * half;
                out[(size_t)(b0 + row) * ODIM + col] = acc[t][r] + ob;
            }
        }
    }
}

extern "C" void kernel_launch(void* const* d_in, const int* in_sizes, int n_in,
                              void* d_out, int out_size, void* d_ws, size_t ws_size,
                              hipStream_t stream) {
    (void)in_sizes; (void)n_in; (void)out_size; (void)ws_size;
    const float* x     = (const float*)d_in[0];
    const float* Wi0   = (const float*)d_in[1];
    const float* Wr0   = (const float*)d_in[2];
    const float* bias0 = (const float*)d_in[3];
    const float* tau0  = (const float*)d_in[4];
    const float* A0    = (const float*)d_in[5];
    const float* m0    = (const float*)d_in[6];
    const float* g0    = (const float*)d_in[7];
    const float* be0   = (const float*)d_in[8];
    const float* Wi1   = (const float*)d_in[9];
    const float* Wr1   = (const float*)d_in[10];
    const float* bias1 = (const float*)d_in[11];
    const float* tau1  = (const float*)d_in[12];
    const float* A1    = (const float*)d_in[13];
    const float* m1    = (const float*)d_in[14];
    const float* g1    = (const float*)d_in[15];
    const float* be1   = (const float*)d_in[16];
    const float* Wi2   = (const float*)d_in[17];
    const float* Wr2   = (const float*)d_in[18];
    const float* bias2 = (const float*)d_in[19];
    const float* tau2  = (const float*)d_in[20];
    const float* A2    = (const float*)d_in[21];
    const float* m2    = (const float*)d_in[22];
    const float* g2    = (const float*)d_in[23];
    const float* be2   = (const float*)d_in[24];
    const float* outW  = (const float*)d_in[25];
    const float* outb  = (const float*)d_in[26];

    __bf16* ws = (__bf16*)d_ws;

    // Pack all weights to bf16 WMMA B-fragment layout (masked/transposed).
    auto pack = [&](const float* src, const float* mask, size_t off,
                    int K, int N, int tr) {
        int total = K * N;
        lnn_pack_b_kernel<<<(total + 255) / 256, 256, 0, stream>>>(
            src, mask, ws + off, K, N, tr);
    };
    pack(Wi0, nullptr, OFF_WI0, 256, 512, 0);
    pack(Wi1, nullptr, OFF_WI1, 512, 512, 0);
    pack(Wi2, nullptr, OFF_WI2, 512, 512, 0);
    pack(Wr0, m0,      OFF_WR0, 512, 512, 1);   // B[k][n] = W_rec[n][k]*mask[n][k]
    pack(Wr1, m1,      OFF_WR1, 512, 512, 1);
    pack(Wr2, m2,      OFF_WR2, 512, 512, 1);
    pack(outW, nullptr, OFF_WO, 512, 128, 1);   // B[k][n] = out_W[n][k]

    hipFuncSetAttribute((const void*)lnn_fused_kernel,
                        hipFuncAttributeMaxDynamicSharedMemorySize, SMEM_BYTES);

    lnn_fused_kernel<<<BATCH / TILE_B, NTHREADS, SMEM_BYTES, stream>>>(
        x,
        bias0, tau0, A0, g0, be0,
        bias1, tau1, A1, g1, be1,
        bias2, tau2, A2, g2, be2,
        outb, ws, (float*)d_out);
}
